// MultiHeadedSelfAttentionModule_21363167330656
// MI455X (gfx1250) — matrix operand order repaired and checked by reading, not compiled, not run on hardware
//
#include <hip/hip_runtime.h>
#include <hip/hip_bf16.h>

#define DM   1024
#define NH   16
#define DH   64
#define BB   8
#define TT   1024
#define PP   (2*TT-1)   /* 2047 */
#define PPAD 16         /* front pad rows of projected P */
#define PROWS 2112      /* padded rows for projected P  */

typedef __attribute__((ext_vector_type(16))) __bf16 v16bf;
typedef __attribute__((ext_vector_type(8)))  __bf16 v8bf;
typedef __attribute__((ext_vector_type(8)))  float  v8f;
typedef __attribute__((ext_vector_type(4)))  unsigned int v4u;
typedef __attribute__((ext_vector_type(8)))  int v8i;
typedef __attribute__((ext_vector_type(4)))  int v4i;

#define DEV __device__ __forceinline__

DEV v8f wmma_bf16(v16bf a, v16bf b, v8f c) {
#if defined(__gfx1250__)
  return __builtin_amdgcn_wmma_f32_16x16x32_bf16(false, a, false, b, (short)0, c, false, false);
#else
  (void)a; (void)b;
  return c;
#endif
}

DEV v8f v8f_zero() {
  v8f z;
  #pragma unroll
  for (int i = 0; i < 8; ++i) z[i] = 0.f;
  return z;
}

#if defined(__gfx1250__) && __has_builtin(__builtin_amdgcn_tensor_load_to_lds)
#define USE_TDM 1
#else
#define USE_TDM 0
#endif

#if USE_TDM
// 2D tile DMA: elements are 2 bytes. Descriptor per CDNA5 ISA D# layout.
// 6-arg builtin form: (g0 v4u, g1 v8i, g2 v4i, g3 v4i, g4 v8i, cpol i32)
DEV void tdm_load_2d(unsigned lds_byte_off, const void* gaddr,
                     unsigned tensor_d0, unsigned tensor_d1,
                     unsigned tile_d0, unsigned tile_d1, unsigned stride0) {
  unsigned long long ga = (unsigned long long)gaddr;
  v4u g0;
  g0[0] = 1u;                                   // count=1, no gather
  g0[1] = lds_byte_off;                         // lds_addr
  g0[2] = (unsigned)ga;                         // global_addr[31:0]
  g0[3] = (unsigned)((ga >> 32) & 0x01FFFFFFu) | 0x80000000u; // addr[56:32] | type=2
  v8i g1;
  g1[0] = (int)(1u << 16);                      // workgroup_mask=0, data_size=1 (2B)
  g1[1] = (int)((tensor_d0 & 0xFFFFu) << 16);   // [47:32]=atomic addr 0, [63:48]=dim0 lo
  g1[2] = (int)(((tensor_d0 >> 16) & 0xFFFFu) | ((tensor_d1 & 0xFFFFu) << 16));
  g1[3] = (int)(((tensor_d1 >> 16) & 0xFFFFu) | ((tile_d0 & 0xFFFFu) << 16));
  g1[4] = (int)(tile_d1 & 0xFFFFu);             // tile_dim1, tile_dim2=0
  g1[5] = (int)stride0;                         // tensor_dim0_stride lo
  g1[6] = 0;                                    // stride0 hi, dim1_stride lo
  g1[7] = 0;
  v4i g2; v4i g3;
  #pragma unroll
  for (int i = 0; i < 4; ++i) { g2[i] = 0; g3[i] = 0; }
  v8i g4;
  #pragma unroll
  for (int i = 0; i < 8; ++i) g4[i] = 0;
  __builtin_amdgcn_tensor_load_to_lds(g0, g1, g2, g3, g4, 0);
}
DEV unsigned lds_off_of(const void* p) { return (unsigned)(unsigned long long)p; }
#endif

// ---------------------------------------------------------------- fp32 -> bf16
__global__ void cvt_bf16_kernel(const float* __restrict__ in, __bf16* __restrict__ out, int n) {
  int i = blockIdx.x * blockDim.x + threadIdx.x;
  if (i < n) out[i] = (__bf16)in[i];
}

// ---------------------------------------------------------------- LayerNorm
__global__ __launch_bounds__(256)
void layernorm_kernel(const float* __restrict__ x, const float* __restrict__ g,
                      const float* __restrict__ beta, __bf16* __restrict__ out) {
  int row = blockIdx.x;
  const float* xr = x + (size_t)row * DM;
  float v[4], s = 0.f, s2 = 0.f;
  #pragma unroll
  for (int i = 0; i < 4; ++i) {
    v[i] = xr[threadIdx.x + 256 * i];
    s += v[i]; s2 += v[i] * v[i];
  }
  #pragma unroll
  for (int m = 16; m; m >>= 1) { s += __shfl_xor(s, m, 32); s2 += __shfl_xor(s2, m, 32); }
  __shared__ float red[16];
  int wid = threadIdx.x >> 5;
  if ((threadIdx.x & 31) == 0) { red[wid] = s; red[8 + wid] = s2; }
  __syncthreads();
  if (threadIdx.x == 0) {
    float a = 0.f, b2 = 0.f;
    for (int i = 0; i < 8; ++i) { a += red[i]; b2 += red[8 + i]; }
    red[0] = a; red[8] = b2;
  }
  __syncthreads();
  float mu   = red[0] * (1.f / DM);
  float var  = red[8] * (1.f / DM) - mu * mu;
  float rstd = rsqrtf(var + 1e-5f);
  #pragma unroll
  for (int i = 0; i < 4; ++i) {
    int c = threadIdx.x + 256 * i;
    out[(size_t)row * DM + c] = (__bf16)((v[i] - mu) * rstd * g[c] + beta[c]);
  }
}

// ---------------------------------------------------------------- pos encoding
__global__ __launch_bounds__(256)
void posemb_kernel(__bf16* __restrict__ pe) {
  int r = blockIdx.x;                       // 0..2046, pos value = T-1-r
  float pos = (float)(TT - 1 - r);
  const float c0 = -9.210340371976184f / (float)DM;   // -ln(10000)/D
  #pragma unroll
  for (int i = 0; i < 4; ++i) {
    int c = threadIdx.x + 256 * i;
    int j = c >> 1;
    float ang = pos * __expf((float)(2 * j) * c0);
    float val = (c & 1) ? __cosf(ang) : __sinf(ang);
    pe[(size_t)r * DM + c] = (__bf16)val;
  }
}

// ---------------------------------------------------------------- GEMM C = A @ W^T (+bias)
// A: MxK bf16 row-major; W: NxK bf16 row-major. 128x128 block, 8 waves, K-step 32.
// TDM path: double-buffered LDS tiles; issue next (A,B) DMA pair, then
// s_wait_tensorcnt 2 -> current pair complete (in-order), next pair in flight.
__global__ __launch_bounds__(256)
void gemm_kernel(const __bf16* __restrict__ A, const __bf16* __restrict__ W,
                 const float* __restrict__ bias, __bf16* __restrict__ outb,
                 float* __restrict__ outf, int M, int N, int K, int f32out) {
  __shared__ __align__(64) __bf16 As[2][128][32];
  __shared__ __align__(64) __bf16 Bs[2][128][32];
  int tid = threadIdx.x;
  int lane = tid & 31, w = tid >> 5;
  int wm = w >> 1, wn = w & 1;          // 4x2 wave grid; wave tile 32x64
  int l15 = lane & 15, half = lane >> 4;
  int m0 = blockIdx.y * 128, n0 = blockIdx.x * 128;
  const int nk = K / 32;

  v8f acc[2][4];
  #pragma unroll
  for (int mt = 0; mt < 2; ++mt)
    #pragma unroll
    for (int nt = 0; nt < 4; ++nt) acc[mt][nt] = v8f_zero();

#if USE_TDM
  if (tid == 0) {   // prologue: stage 0 into buffer 0
    tdm_load_2d(lds_off_of(&As[0][0][0]), A + (size_t)m0 * K,
                (unsigned)K, (unsigned)(M - m0), 32u, 128u, (unsigned)K);
    tdm_load_2d(lds_off_of(&Bs[0][0][0]), W + (size_t)n0 * K,
                (unsigned)K, (unsigned)(N - n0), 32u, 128u, (unsigned)K);
  }
#endif

  for (int kt = 0; kt < nk; ++kt) {
    int cur = kt & 1;
    int k0 = kt * 32;
#if USE_TDM
    if (tid == 0 && kt + 1 < nk) {      // prefetch next stage into other buffer
      int k1 = k0 + 32;
      tdm_load_2d(lds_off_of(&As[cur ^ 1][0][0]), A + (size_t)m0 * K + k1,
                  (unsigned)(K - k1), (unsigned)(M - m0), 32u, 128u, (unsigned)K);
      tdm_load_2d(lds_off_of(&Bs[cur ^ 1][0][0]), W + (size_t)n0 * K + k1,
                  (unsigned)(K - k1), (unsigned)(N - n0), 32u, 128u, (unsigned)K);
    }
    if (tid < 32) {                     // immediates only for s_wait_tensorcnt
      if (kt + 1 < nk) __builtin_amdgcn_s_wait_tensorcnt(2);
      else             __builtin_amdgcn_s_wait_tensorcnt(0);
    }
#else
    {
      int r = tid >> 1, seg = tid & 1;
      int gr = m0 + r;
      if (gr < M) {
        const __bf16* src = A + (size_t)gr * K + k0 + seg * 16;
        *(v8bf*)&As[cur][r][seg * 16]     = *(const v8bf*)src;
        *(v8bf*)&As[cur][r][seg * 16 + 8] = *(const v8bf*)(src + 8);
      } else {
        #pragma unroll
        for (int e = 0; e < 16; ++e) As[cur][r][seg * 16 + e] = (__bf16)0.f;
      }
      int gn = n0 + r;
      if (gn < N) {
        const __bf16* src = W + (size_t)gn * K + k0 + seg * 16;
        *(v8bf*)&Bs[cur][r][seg * 16]     = *(const v8bf*)src;
        *(v8bf*)&Bs[cur][r][seg * 16 + 8] = *(const v8bf*)(src + 8);
      } else {
        #pragma unroll
        for (int e = 0; e < 16; ++e) Bs[cur][r][seg * 16 + e] = (__bf16)0.f;
      }
      if (k0 + 32 < K && gr < M)
        __builtin_prefetch(A + (size_t)gr * K + k0 + 32, 0, 3);
    }
#endif
    __syncthreads();                    // tiles[cur] ready for all waves

    v16bf afrag[2];
    #pragma unroll
    for (int mt = 0; mt < 2; ++mt) {
      int r = wm * 32 + mt * 16 + l15;
      v8bf lo = *(const v8bf*)&As[cur][r][half * 16];
      v8bf hi = *(const v8bf*)&As[cur][r][half * 16 + 8];
      afrag[mt] = __builtin_shufflevector(lo, hi, 0,1,2,3,4,5,6,7,8,9,10,11,12,13,14,15);
    }
    #pragma unroll
    for (int nt = 0; nt < 4; ++nt) {
      int cn = wn * 64 + nt * 16 + l15;
      v16bf bfrag = *(const v16bf*)&Bs[cur][cn][half * 16];
      #pragma unroll
      for (int mt = 0; mt < 2; ++mt)
        acc[mt][nt] = wmma_bf16(afrag[mt], bfrag, acc[mt][nt]);
    }
    __syncthreads();                    // all waves done with tiles[cur]
  }

  #pragma unroll
  for (int mt = 0; mt < 2; ++mt) {
    #pragma unroll
    for (int nt = 0; nt < 4; ++nt) {
      int colg = n0 + wn * 64 + nt * 16 + l15;
      float bs = bias ? bias[colg] : 0.f;
      #pragma unroll
      for (int i = 0; i < 8; ++i) {
        int rowg = m0 + wm * 32 + mt * 16 + i + 8 * half;
        if (rowg < M) {
          float vv = acc[mt][nt][i] + bs;
          if (f32out) outf[(size_t)rowg * N + colg] = vv;
          else        outb[(size_t)rowg * N + colg] = (__bf16)vv;
        }
      }
    }
  }
}

// ---------------------------------------------------------------- fused rel-pos flash attention
// block = (b, h, 128 query rows); wave w owns 16 rows; loop over 64-key blocks.
__global__ __launch_bounds__(256)
void attention_kernel(const __bf16* __restrict__ q, const __bf16* __restrict__ k,
                      const __bf16* __restrict__ v, const __bf16* __restrict__ p, // padded base
                      const float* __restrict__ ub, const float* __restrict__ vb,
                      __bf16* __restrict__ ctx) {
  __shared__ __align__(64) __bf16 Vt[64][64];     // V^T tile: Vt[d][s]   8 KB
  __shared__ float band[8][16][80];               // per-wave pos band   40 KB
  __shared__ __align__(64) __bf16 Pl[8][16][64];  // per-wave P relayout 16 KB

  int tid = threadIdx.x, lane = tid & 31, w = tid >> 5;
  int l15 = lane & 15, half = lane >> 4;
  int bid = blockIdx.x;
  int t0 = (bid & 7) * 128;
  int h  = (bid >> 3) & 15;
  int b  = bid >> 7;
  int tw = t0 + w * 16;

  // Q fragments with u/v biases folded in (A-fragment layout, dh=64 -> 2 K-steps)
  v16bf qc[2], qp[2];
  {
    const __bf16* qrow = q + ((size_t)(b * TT + tw + l15)) * DM + h * DH;
    const float* uh = ub + h * DH;
    const float* vh = vb + h * DH;
    #pragma unroll
    for (int ks = 0; ks < 2; ++ks) {
      #pragma unroll
      for (int e = 0; e < 8; ++e) {
        int d0 = ks * 32 + half * 16 + e;
        int d1 = ks * 32 + half * 16 + 8 + e;
        float f0 = (float)qrow[d0], f1 = (float)qrow[d1];
        qc[ks][e]     = (__bf16)(f0 + uh[d0]);
        qc[ks][8 + e] = (__bf16)(f1 + uh[d1]);
        qp[ks][e]     = (__bf16)(f0 + vh[d0]);
        qp[ks][8 + e] = (__bf16)(f1 + vh[d1]);
      }
    }
  }

  float mrow[8], lrow[8];
  v8f ctxacc[4];
  #pragma unroll
  for (int i = 0; i < 8; ++i) { mrow[i] = -1e30f; lrow[i] = 0.f; }
  #pragma unroll
  for (int dt = 0; dt < 4; ++dt) ctxacc[dt] = v8f_zero();

  int base = TT - 1 - tw;   // abs p index of (t=tw, s=0) diagonal

  for (int s0 = 0; s0 < TT; s0 += 64) {
    __syncthreads();
    { // cooperative V^T staging (coalesced global reads, transposed LDS writes)
      int s = tid >> 2, dseg = tid & 3;
      const __bf16* vrow = v + ((size_t)(b * TT + s0 + s)) * DM + h * DH + dseg * 16;
      #pragma unroll
      for (int j = 0; j < 16; ++j) Vt[dseg * 16 + j][s] = vrow[j];
    }
    __syncthreads();

    if (s0 + 64 < TT)
      __builtin_prefetch(k + ((size_t)(b * TT + s0 + 64 + l15)) * DM + h * DH, 0, 3);

    // content scores: 4 tiles of 16x16
    v8f sc[4];
    #pragma unroll
    for (int nt = 0; nt < 4; ++nt) {
      int scol = s0 + nt * 16 + l15;
      const __bf16* krow = k + ((size_t)(b * TT + scol)) * DM + h * DH;
      v16bf kb0 = *(const v16bf*)(krow + half * 16);
      v16bf kb1 = *(const v16bf*)(krow + 32 + half * 16);
      v8f a = v8f_zero();
      a = wmma_bf16(qc[0], kb0, a);
      a = wmma_bf16(qc[1], kb1, a);
      sc[nt] = a;
    }

    // pos band: 5 tiles covering abs p cols [base+s0-16, base+s0+64)
    #pragma unroll
    for (int bt = 0; bt < 5; ++bt) {
      int pidx = base + s0 - 16 + bt * 16 + l15 + PPAD;   // padded row index
      const __bf16* prow = p + (size_t)pidx * DM + h * DH;
      v16bf pb0 = *(const v16bf*)(prow + half * 16);
      v16bf pb1 = *(const v16bf*)(prow + 32 + half * 16);
      v8f a = v8f_zero();
      a = wmma_bf16(qp[0], pb0, a);
      a = wmma_bf16(qp[1], pb1, a);
      #pragma unroll
      for (int i = 0; i < 8; ++i) band[w][i + 8 * half][bt * 16 + l15] = a[i];
    }

    // combine + diagonal shift-gather + online softmax
    float sv[4][8], mx[8];
    #pragma unroll
    for (int i = 0; i < 8; ++i) mx[i] = -1e30f;
    #pragma unroll
    for (int nt = 0; nt < 4; ++nt) {
      #pragma unroll
      for (int i = 0; i < 8; ++i) {
        int row = i + 8 * half;
        float pos = band[w][row][16 + nt * 16 + l15 - row];  // in [1,79]
        float xv = 0.125f * (sc[nt][i] + pos);               // 1/sqrt(dh)
        sv[nt][i] = xv;
        mx[i] = fmaxf(mx[i], xv);
      }
    }
    #pragma unroll
    for (int i = 0; i < 8; ++i) {
      #pragma unroll
      for (int msk = 8; msk; msk >>= 1) mx[i] = fmaxf(mx[i], __shfl_xor(mx[i], msk, 32));
      float nm = fmaxf(mrow[i], mx[i]);
      float alpha = __expf(mrow[i] - nm);
      mrow[i] = nm;
      float rs = 0.f;
      #pragma unroll
      for (int nt = 0; nt < 4; ++nt) {
        float pv = __expf(sv[nt][i] - nm);
        sv[nt][i] = pv;
        rs += pv;
      }
      #pragma unroll
      for (int msk = 8; msk; msk >>= 1) rs += __shfl_xor(rs, msk, 32);
      lrow[i] = lrow[i] * alpha + rs;
      #pragma unroll
      for (int dt = 0; dt < 4; ++dt) ctxacc[dt][i] *= alpha;
    }

    // P tile (C-layout) -> LDS bf16 -> A-fragment layout
    #pragma unroll
    for (int nt = 0; nt < 4; ++nt)
      #pragma unroll
      for (int i = 0; i < 8; ++i)
        Pl[w][i + 8 * half][nt * 16 + l15] = (__bf16)sv[nt][i];

    // ctx += P(16x64) @ V(64x64)
    #pragma unroll
    for (int ks2 = 0; ks2 < 2; ++ks2) {
      v8bf lo = *(const v8bf*)&Pl[w][l15][ks2 * 32 + half * 16];
      v8bf hi = *(const v8bf*)&Pl[w][l15][ks2 * 32 + half * 16 + 8];
      v16bf afrag = __builtin_shufflevector(lo, hi, 0,1,2,3,4,5,6,7,8,9,10,11,12,13,14,15);
      #pragma unroll
      for (int dt = 0; dt < 4; ++dt) {
        v16bf bfrag = *(const v16bf*)&Vt[dt * 16 + l15][ks2 * 32 + half * 16];
        ctxacc[dt] = wmma_bf16(afrag, bfrag, ctxacc[dt]);
      }
    }
  }

  #pragma unroll
  for (int i = 0; i < 8; ++i) {
    float inv = 1.f / lrow[i];
    int trow = tw + i + 8 * half;
    #pragma unroll
    for (int dt = 0; dt < 4; ++dt) {
      int col = h * DH + dt * 16 + l15;
      ctx[((size_t)(b * TT + trow)) * DM + col] = (__bf16)(ctxacc[dt][i] * inv);
    }
  }
}

// ---------------------------------------------------------------- launcher
extern "C" void kernel_launch(void* const* d_in, const int* in_sizes, int n_in,
                              void* d_out, int out_size, void* d_ws, size_t ws_size,
                              hipStream_t stream) {
  (void)in_sizes; (void)n_in; (void)out_size; (void)ws_size;
  const float* x    = (const float*)d_in[0];
  const float* Wq   = (const float*)d_in[1];
  const float* bq   = (const float*)d_in[2];
  const float* Wk   = (const float*)d_in[3];
  const float* bk   = (const float*)d_in[4];
  const float* Wv   = (const float*)d_in[5];
  const float* bv   = (const float*)d_in[6];
  const float* Wpos = (const float*)d_in[7];
  const float* Wout = (const float*)d_in[8];
  const float* bout = (const float*)d_in[9];
  const float* ub   = (const float*)d_in[10];
  const float* vb   = (const float*)d_in[11];
  const float* g    = (const float*)d_in[12];
  const float* be   = (const float*)d_in[13];
  float* out = (float*)d_out;

  char* ws = (char*)d_ws;
  size_t off = 0;
  auto wsalloc = [&](size_t bytes) -> void* {
    void* pr = ws + off; off += (bytes + 255) & ~(size_t)255; return pr;
  };
  __bf16* xn   = (__bf16*)wsalloc((size_t)BB * TT * DM * 2);
  __bf16* wqb  = (__bf16*)wsalloc((size_t)DM * DM * 2);
  __bf16* wkb  = (__bf16*)wsalloc((size_t)DM * DM * 2);
  __bf16* wvb  = (__bf16*)wsalloc((size_t)DM * DM * 2);
  __bf16* wpb  = (__bf16*)wsalloc((size_t)DM * DM * 2);
  __bf16* wob  = (__bf16*)wsalloc((size_t)DM * DM * 2);
  __bf16* pe   = (__bf16*)wsalloc((size_t)PP * DM * 2);
  __bf16* qb   = (__bf16*)wsalloc((size_t)BB * TT * DM * 2);
  __bf16* kb   = (__bf16*)wsalloc((size_t)BB * TT * DM * 2);
  __bf16* vbq  = (__bf16*)wsalloc((size_t)BB * TT * DM * 2);
  __bf16* ppad = (__bf16*)wsalloc((size_t)PROWS * DM * 2);
  __bf16* ctx  = (__bf16*)wsalloc((size_t)BB * TT * DM * 2);

  const int nW = DM * DM;
  dim3 cg((nW + 255) / 256);
  cvt_bf16_kernel<<<cg, 256, 0, stream>>>(Wq,   wqb, nW);
  cvt_bf16_kernel<<<cg, 256, 0, stream>>>(Wk,   wkb, nW);
  cvt_bf16_kernel<<<cg, 256, 0, stream>>>(Wv,   wvb, nW);
  cvt_bf16_kernel<<<cg, 256, 0, stream>>>(Wpos, wpb, nW);
  cvt_bf16_kernel<<<cg, 256, 0, stream>>>(Wout, wob, nW);

  (void)hipMemsetAsync(ppad, 0, (size_t)PROWS * DM * 2, stream);

  layernorm_kernel<<<BB * TT, 256, 0, stream>>>(x, g, be, xn);
  posemb_kernel<<<PP, 256, 0, stream>>>(pe);

  dim3 gg(DM / 128, (BB * TT) / 128);
  gemm_kernel<<<gg, 256, 0, stream>>>(xn, wqb, bq, qb,  nullptr, BB * TT, DM, DM, 0);
  gemm_kernel<<<gg, 256, 0, stream>>>(xn, wkb, bk, kb,  nullptr, BB * TT, DM, DM, 0);
  gemm_kernel<<<gg, 256, 0, stream>>>(xn, wvb, bv, vbq, nullptr, BB * TT, DM, DM, 0);
  dim3 gp(DM / 128, (PP + 127) / 128);
  gemm_kernel<<<gp, 256, 0, stream>>>(pe, wpb, nullptr, ppad + (size_t)PPAD * DM,
                                      nullptr, PP, DM, DM, 0);

  attention_kernel<<<BB * NH * (TT / 128), 256, 0, stream>>>(qb, kb, vbq, ppad, ub, vb, ctx);

  gemm_kernel<<<gg, 256, 0, stream>>>(ctx, wob, bout, nullptr, out, BB * TT, DM, DM, 1);
}
